// CapsNet_90847148245743
// MI455X (gfx1250) — matrix operand order repaired and checked by reading, not compiled, not run on hardware
//
#include <hip/hip_runtime.h>
#include <hip/hip_bf16.h>

typedef _Float16 f16;
typedef _Float16 v16h __attribute__((ext_vector_type(16)));
typedef _Float16 v8h  __attribute__((ext_vector_type(8)));
typedef float    v8f  __attribute__((ext_vector_type(8)));

#define BB   16384
#define HH   136
#define LL   272
#define NLAY 3
#define KP1  288   // 272 padded to 9 k-tiles of 32
#define KPH  160   // 136 padded to 5 k-tiles of 32
#define N1   1088  // 2 dirs * 4H
#define NHG  544   // 4H

__device__ __forceinline__ float sigf(float x) { return 1.0f / (1.0f + expf(-x)); }

__device__ __forceinline__ float wred(float v) {
    #pragma unroll
    for (int off = 16; off > 0; off >>= 1) v += __shfl_xor(v, off, 32);
    return v;
}

// ---------------- weight prep: cast + zero-pad K dimension ----------------
__global__ void k_cast_pad(const float* __restrict__ src, f16* __restrict__ dst,
                           int rows, int K, int Kp) {
    long idx = (long)blockIdx.x * blockDim.x + threadIdx.x;
    long total = (long)rows * Kp;
    if (idx >= total) return;
    int r = (int)(idx / Kp);
    int k = (int)(idx - (long)r * Kp);
    dst[idx] = (f16)(k < K ? src[(size_t)r * K + k] : 0.0f);
}

__global__ void k_bias_sum(const float* __restrict__ bi, const float* __restrict__ bh,
                           float* __restrict__ out, int n) {
    int i = blockIdx.x * blockDim.x + threadIdx.x;
    if (i < n) out[i] = bi[i] + bh[i];
}

// ---------------- build x: timestep0 = embedding, timestep1 = signal ----------------
__global__ void k_embed(const int* __restrict__ seq, const float* __restrict__ sig,
                        const float* __restrict__ embed, f16* __restrict__ X16) {
    int b = blockIdx.x;
    int k = threadIdx.x;                 // blockDim.x == 288
    float v0 = 0.0f, v1 = 0.0f;
    if (k < LL) {
        int j = k >> 4, e = k & 15;
        v0 = embed[seq[b * 17 + j] * 16 + e];
        v1 = sig[(size_t)b * LL + k];
    }
    X16[(size_t)b * KP1 + k]            = (f16)v0;   // t=0 row
    X16[(size_t)(BB + b) * KP1 + k]     = (f16)v1;   // t=1 row
}

// ---------------- WMMA GEMM: C[M,N] = A[M,K] * W[N,K]^T (+bias) ----------------
// block = (32,8): 8 waves; each wave owns a 32(M) x (NT*16)(N) output tile.
// WG covers 256 M rows. NT=4 for the big input GEMM (N1 = 17*64), NT=2 for the
// recurrent GEMM (NHG = 17*32). Single base pointers + uniform offsets keep all
// loads in the GLOBAL address space (no flat_load / DScnt coupling).
template <int NT>
__global__ __launch_bounds__(256)
void k_gemm(const f16* __restrict__ A, int lda, long strideAz,
            const f16* __restrict__ W, int ldw, long strideWz,
            const float* __restrict__ bias, long strideBz,
            float* __restrict__ C, int ldc, long strideCz, int KT) {
    int lane = threadIdx.x;
    int wv   = threadIdx.y;
    size_t z = blockIdx.z;
    A += z * (size_t)strideAz;
    W += z * (size_t)strideWz;
    C += z * (size_t)strideCz;
    if (bias) bias += z * (size_t)strideBz;

    int m0 = (blockIdx.y * 8 + wv) * 32;
    int n0 = blockIdx.x * (16 * NT);
    int lm = lane & 15;
    int hi = lane >> 4;                       // 0 | 1

    // A lane layout (16-bit 16x32): halfs[0:16] = K[hi*16 .. +15] (16 contiguous halfs)
    const f16* ap = A + (size_t)(m0 + lm) * lda + hi * 16;
    const size_t astep = (size_t)16 * lda;
    // B lane layout (32x16): lane n holds W[n0+n][kb + hi*16 .. +15] (16 contiguous halfs)
    const f16* bp = W + (size_t)(n0 + lm) * ldw + hi * 16;
    const size_t bstep = (size_t)16 * ldw;

    v8f acc0[NT], acc1[NT];
    #pragma unroll
    for (int t = 0; t < NT; ++t) { acc0[t] = (v8f){}; acc1[t] = (v8f){}; }

    union AV { v16h v; v8h h[2]; };

    for (int kt = 0; kt < KT; ++kt) {
        AV a0, a1;
        a0.h[0] = *(const v8h*)(ap);                 a0.h[1] = *(const v8h*)(ap + 8);
        a1.h[0] = *(const v8h*)(ap + astep);         a1.h[1] = *(const v8h*)(ap + astep + 8);
        v16h bt[NT];
        #pragma unroll
        for (int t = 0; t < NT; ++t) bt[t] = *(const v16h*)(bp + (size_t)t * bstep);
        #pragma unroll
        for (int t = 0; t < NT; ++t) {
            acc0[t] = __builtin_amdgcn_wmma_f32_16x16x32_f16(false, a0.v, false, bt[t], (short)0, acc0[t], false, false);
            acc1[t] = __builtin_amdgcn_wmma_f32_16x16x32_f16(false, a1.v, false, bt[t], (short)0, acc1[t], false, false);
        }
        ap += 32; bp += 32;
    }

    if (bias) {
        #pragma unroll
        for (int t = 0; t < NT; ++t) {
            float bv = bias[n0 + t * 16 + lm];
            #pragma unroll
            for (int r = 0; r < 8; ++r) { acc0[t][r] += bv; acc1[t][r] += bv; }
        }
    }

    int row0 = m0 + hi * 8;        // D layout: VGPR r -> row r (lanes 0-15) / row r+8 (lanes 16-31)
    int row1 = m0 + 16 + hi * 8;
    #pragma unroll
    for (int t = 0; t < NT; ++t) {
        int nc = n0 + t * 16 + lm;
        #pragma unroll
        for (int r = 0; r < 8; ++r) {
            C[(size_t)(row0 + r) * ldc + nc] = acc0[t][r];
            C[(size_t)(row1 + r) * ldc + nc] = acc1[t][r];
        }
    }
}

// ---------------- LSTM cell 1: first step of each direction (state = 0) ----------------
__global__ void k_cell1(const float* __restrict__ G,
                        float* __restrict__ hf0, float* __restrict__ cf,
                        float* __restrict__ hb1, float* __restrict__ cb,
                        f16* __restrict__ A16) {
    int idx = blockIdx.x * blockDim.x + threadIdx.x;
    if (idx >= BB * HH) return;
    int b = idx / HH, j = idx - b * HH;

    // forward dir, t=0: G row b, gates [0,544)
    const float* g = G + (size_t)b * N1;
    float c = sigf(g[j]) * tanhf(g[2 * HH + j]);
    float h = sigf(g[3 * HH + j]) * tanhf(c);
    hf0[idx] = h; cf[idx] = c;
    A16[(size_t)b * KPH + j] = (f16)h;

    // backward dir, t=1: G row BB+b, gates [544,1088)
    g = G + (size_t)(BB + b) * N1 + NHG;
    c = sigf(g[j]) * tanhf(g[2 * HH + j]);
    h = sigf(g[3 * HH + j]) * tanhf(c);
    hb1[idx] = h; cb[idx] = c;
    A16[(size_t)(BB + b) * KPH + j] = (f16)h;

    if (j < KPH - HH) {  // zero the K padding every call (ws is poisoned)
        A16[(size_t)b * KPH + HH + j]        = (f16)0.0f;
        A16[(size_t)(BB + b) * KPH + HH + j] = (f16)0.0f;
    }
}

// ---------------- LSTM cell 2: second step + assemble next-layer x ----------------
__global__ void k_cell2(const float* __restrict__ G, const float* __restrict__ R,
                        const float* __restrict__ hf0, const float* __restrict__ cf,
                        const float* __restrict__ hb1, const float* __restrict__ cb,
                        f16* __restrict__ X16, float* __restrict__ X32) {
    int idx = blockIdx.x * blockDim.x + threadIdx.x;
    if (idx >= BB * HH) return;
    int b = idx / HH, j = idx - b * HH;

    // hf1: forward dir, t=1
    const float* g = G + (size_t)(BB + b) * N1;
    const float* r = R + (size_t)b * NHG;
    float i_ = g[j] + r[j];
    float f_ = g[HH + j] + r[HH + j];
    float gg = g[2 * HH + j] + r[2 * HH + j];
    float o_ = g[3 * HH + j] + r[3 * HH + j];
    float c  = sigf(f_) * cf[idx] + sigf(i_) * tanhf(gg);
    float hf1 = sigf(o_) * tanhf(c);

    // hb0: backward dir, t=0
    g = G + (size_t)b * N1 + NHG;
    r = R + (size_t)(BB + b) * NHG;
    i_ = g[j] + r[j];
    f_ = g[HH + j] + r[HH + j];
    gg = g[2 * HH + j] + r[2 * HH + j];
    o_ = g[3 * HH + j] + r[3 * HH + j];
    c  = sigf(f_) * cb[idx] + sigf(i_) * tanhf(gg);
    float hb0 = sigf(o_) * tanhf(c);

    float h0 = hf0[idx], h1b = hb1[idx];
    size_t x0 = (size_t)b * KP1, x1 = (size_t)(BB + b) * KP1;
    X16[x0 + j]      = (f16)h0;   X16[x0 + HH + j] = (f16)hb0;
    X16[x1 + j]      = (f16)hf1;  X16[x1 + HH + j] = (f16)h1b;
    size_t y0 = (size_t)b * LL, y1 = (size_t)(BB + b) * LL;
    X32[y0 + j] = h0;  X32[y0 + HH + j] = hb0;
    X32[y1 + j] = hf1; X32[y1 + HH + j] = h1b;
}

// ---------------- head: conv + squash + u_hat + (degenerate) routing + FCs ----------------
// One wave per batch element, 4 elements per 128-thread block.
__global__ __launch_bounds__(128)
void k_head(const float* __restrict__ X32, const float* __restrict__ conv_w,
            const float* __restrict__ conv_b, const float* __restrict__ W_caps,
            const float* __restrict__ fc1_w, const float* __restrict__ fc1_b,
            const float* __restrict__ fc2_w, const float* __restrict__ fc2_b,
            float* __restrict__ out) {
    __shared__ float u[4][10][LL];
    __shared__ float xs[4][2][LL];
    __shared__ float uh[4][10][16];
    int lane = threadIdx.x & 31;
    int w    = threadIdx.x >> 5;
    int b    = blockIdx.x * 4 + w;

    for (int k = lane; k < LL; k += 32) {
        xs[w][0][k] = X32[(size_t)b * LL + k];
        xs[w][1][k] = X32[(size_t)(BB + b) * LL + k];
    }
    __syncthreads();

    // dilated convs: y[c][t] = b_c + sum_i w[c,i,0]*x_i[t-left] + w[c,i,1]*x_i[t-left+d]
    for (int c = 0; c < 10; ++c) {
        int d = (c >> 1) + 1, left = d >> 1;
        const float* wp = conv_w + (size_t)c * 4;
        float w00 = wp[0], w01 = wp[1], w10 = wp[2], w11 = wp[3];
        float bc = conv_b[c];
        for (int k = lane; k < LL; k += 32) {
            int kl = k - left, kr = k - left + d;
            float x0l = (kl >= 0) ? xs[w][0][kl] : 0.0f;
            float x1l = (kl >= 0) ? xs[w][1][kl] : 0.0f;
            float x0r = (kr < LL) ? xs[w][0][kr] : 0.0f;
            float x1r = (kr < LL) ? xs[w][1][kr] : 0.0f;
            u[w][c][k] = bc + w00 * x0l + w01 * x0r + w10 * x1l + w11 * x1r;
        }
    }
    __syncthreads();

    // per-capsule norm scaling: u *= (1 - 1/(e^n+eps)) / (n+eps)
    for (int c = 0; c < 10; ++c) {
        float p = 0.0f;
        for (int k = lane; k < LL; k += 32) { float t = u[w][c][k]; p += t * t; }
        p = wred(p);
        float n = sqrtf(p);
        float scale = (1.0f - 1.0f / (expf(n) + 1e-20f)) / (n + 1e-20f);
        for (int k = lane; k < LL; k += 32) u[w][c][k] *= scale;
    }
    __syncthreads();

    // u_hat[c][o] = sum_k W_caps[c,o,k] * u[c][k]
    for (int c = 0; c < 10; ++c)
        for (int o = 0; o < 16; ++o) {
            const float* wc = W_caps + ((size_t)c * 16 + o) * LL;
            float p = 0.0f;
            for (int k = lane; k < LL; k += 32) p += wc[k] * u[w][c][k];
            p = wred(p);
            if (lane == 0) uh[w][c][o] = p;
        }
    __syncthreads();

    // routing: softmax over a size-1 axis -> c == 1 always, so v = squash(sum_c u_hat)
    float s[16];
    #pragma unroll
    for (int o = 0; o < 16; ++o) {
        float t = 0.0f;
        for (int c = 0; c < 10; ++c) t += uh[w][c][o];
        s[o] = t;
    }
    float sq = 0.0f;
    #pragma unroll
    for (int o = 0; o < 16; ++o) sq += s[o] * s[o];
    float coef = sq / (1.0f + sq) / sqrtf(sq);
    float v[16];
    #pragma unroll
    for (int o = 0; o < 16; ++o) v[o] = coef * s[o];

    // FC head
    float p0 = 0.0f, p1 = 0.0f;
    for (int j = lane; j < HH; j += 32) {
        float h = fc1_b[j];
        #pragma unroll
        for (int o = 0; o < 16; ++o) h += v[o] * fc1_w[j * 16 + o];
        h = fmaxf(h, 0.0f);
        p0 += h * fc2_w[j];
        p1 += h * fc2_w[HH + j];
    }
    p0 = wred(p0); p1 = wred(p1);
    if (lane == 0) {
        float l0 = p0 + fc2_b[0], l1 = p1 + fc2_b[1];
        out[(size_t)b * 2]     = l0;
        out[(size_t)b * 2 + 1] = l1;
        float m = fmaxf(l0, l1);
        float e0 = expf(l0 - m), e1 = expf(l1 - m);
        float inv = 1.0f / (e0 + e1);
        out[(size_t)BB * 2 + b * 2]     = e0 * inv;
        out[(size_t)BB * 2 + b * 2 + 1] = e1 * inv;
    }
}

extern "C" void kernel_launch(void* const* d_in, const int* in_sizes, int n_in,
                              void* d_out, int out_size, void* d_ws, size_t ws_size,
                              hipStream_t stream) {
    const int* seq      = (const int*)d_in[0];
    const float* sig    = (const float*)d_in[1];
    const float* embed  = (const float*)d_in[2];
    const float* w_ih   = (const float*)d_in[3];
    const float* w_hh   = (const float*)d_in[4];
    const float* b_ih   = (const float*)d_in[5];
    const float* b_hh   = (const float*)d_in[6];
    const float* conv_w = (const float*)d_in[7];
    const float* conv_b = (const float*)d_in[8];
    const float* W_caps = (const float*)d_in[9];
    const float* fc1_w  = (const float*)d_in[10];
    const float* fc1_b  = (const float*)d_in[11];
    const float* fc2_w  = (const float*)d_in[12];
    const float* fc2_b  = (const float*)d_in[13];
    float* out = (float*)d_out;

    char* ws = (char*)d_ws;
    size_t off = 0;
    auto alloc = [&](size_t bytes) -> char* {
        char* p = ws + off;
        off += (bytes + 255) & ~(size_t)255;
        return p;
    };
    const int WROWS = NLAY * 2 * 4 * HH;                 // 3264 gate rows total
    f16*   Wi16 = (f16*)  alloc((size_t)WROWS * KP1 * sizeof(f16));
    f16*   Wh16 = (f16*)  alloc((size_t)WROWS * KPH * sizeof(f16));
    float* bsum = (float*)alloc((size_t)WROWS * sizeof(float));
    f16*   X16  = (f16*)  alloc((size_t)2 * BB * KP1 * sizeof(f16));
    float* X32  = (float*)alloc((size_t)2 * BB * LL * sizeof(float));
    float* G    = (float*)alloc((size_t)2 * BB * N1 * sizeof(float));
    float* R    = (float*)alloc((size_t)2 * BB * NHG * sizeof(float));
    f16*   A16  = (f16*)  alloc((size_t)2 * BB * KPH * sizeof(f16));
    float* hf0  = (float*)alloc((size_t)BB * HH * sizeof(float));
    float* cf   = (float*)alloc((size_t)BB * HH * sizeof(float));
    float* hb1  = (float*)alloc((size_t)BB * HH * sizeof(float));
    float* cb   = (float*)alloc((size_t)BB * HH * sizeof(float));

    // weight prep (tiny, re-done every call for determinism)
    {
        long t1 = (long)WROWS * KP1;
        k_cast_pad<<<(unsigned)((t1 + 255) / 256), 256, 0, stream>>>(w_ih, Wi16, WROWS, LL, KP1);
        long t2 = (long)WROWS * KPH;
        k_cast_pad<<<(unsigned)((t2 + 255) / 256), 256, 0, stream>>>(w_hh, Wh16, WROWS, HH, KPH);
        k_bias_sum<<<(WROWS + 255) / 256, 256, 0, stream>>>(b_ih, b_hh, bsum, WROWS);
    }

    // build x from embedding + signal
    k_embed<<<BB, KP1, 0, stream>>>(seq, sig, embed, X16);

    dim3 blk(32, 8);
    int cellGrid = (BB * HH + 127) / 128;
    for (int l = 0; l < NLAY; ++l) {
        const f16* WiL = Wi16 + (size_t)l * 2 * NHG * KP1;
        const f16* WhL = Wh16 + (size_t)l * 2 * NHG * KPH;
        const float* bL = bsum + (size_t)l * 2 * NHG;
        // G = X * [WiF;WiB]^T + (bi+bh)   -> (2B x 1088), wave tile 32x64
        k_gemm<4><<<dim3(N1 / 64, (2 * BB) / 256, 1), blk, 0, stream>>>(
            X16, KP1, 0, WiL, KP1, 0, bL, 0, G, N1, 0, KP1 / 32);
        k_cell1<<<cellGrid, 128, 0, stream>>>(G, hf0, cf, hb1, cb, A16);
        // R[z] = h[z] * Wh[z]^T  (z=0: hf0/WhF, z=1: hb1/WhB) -> (B x 544), wave tile 32x32
        k_gemm<2><<<dim3(NHG / 32, BB / 256, 2), blk, 0, stream>>>(
            A16, KPH, (long)BB * KPH, WhL, KPH, (long)NHG * KPH, nullptr, 0,
            R, NHG, (long)BB * NHG, KPH / 32);
        k_cell2<<<cellGrid, 128, 0, stream>>>(G, R, hf0, cf, hb1, cb, X16, X32);
    }

    k_head<<<BB / 4, 128, 0, stream>>>(X32, conv_w, conv_b, W_caps,
                                       fc1_w, fc1_b, fc2_w, fc2_b, out);
}
